// GraphNeuralODEPrompt_34248069218344
// MI455X (gfx1250) — matrix operand order repaired and checked by reading, not compiled, not run on hardware
//
#include <hip/hip_runtime.h>
#include <hip/hip_bf16.h>

#define DIM        128
#define NANCH      5
#define ODE_STEPS  5
#define LSTR       144   // padded LDS row stride in halves (288B, 16B aligned)

typedef __attribute__((ext_vector_type(16))) _Float16 v16h;
typedef __attribute__((ext_vector_type(8)))  _Float16 v8h;
typedef __attribute__((ext_vector_type(8)))  float    v8f;

static __device__ __forceinline__ float wave_reduce_sum(float v) {
    v += __shfl_xor(v, 16, 32);
    v += __shfl_xor(v, 8, 32);
    v += __shfl_xor(v, 4, 32);
    v += __shfl_xor(v, 2, 32);
    v += __shfl_xor(v, 1, 32);
    return v;
}

// 16-byte async DMA: global -> LDS, tracked by ASYNCcnt (no VGPR round trip)
static __device__ __forceinline__ void async_copy_b128(uint32_t lds_off,
                                                       const void* gptr) {
    asm volatile("global_load_async_to_lds_b128 %0, %1, off"
                 :: "v"(lds_off), "v"(gptr)
                 : "memory");
}
static __device__ __forceinline__ void wait_async0() {
    asm volatile("s_wait_asynccnt 0" ::: "memory");
}

// ---------------------------------------------------------------------------
// Kernel A: node_x = x + softmax(x @ W_att^T + b_att) @ node_anchor
// one wave (32 lanes) per node, 4 floats per lane
// ---------------------------------------------------------------------------
__global__ __launch_bounds__(256) void node_prompt_kernel(
    const float* __restrict__ x, const float* __restrict__ W_att,
    const float* __restrict__ b_att, const float* __restrict__ node_anchor,
    float* __restrict__ node_x, int N)
{
    const int wave = threadIdx.x >> 5;
    const int lane = threadIdx.x & 31;
    const int node = blockIdx.x * 8 + wave;
    if (node >= N) return;

    const float4 xs = *(const float4*)(x + (size_t)node * DIM + lane * 4);

    float logits[NANCH];
#pragma unroll
    for (int a = 0; a < NANCH; ++a) {
        const float4 w = *(const float4*)(W_att + a * DIM + lane * 4);
        float p = xs.x * w.x + xs.y * w.y + xs.z * w.z + xs.w * w.w;
        logits[a] = wave_reduce_sum(p) + b_att[a];
    }
    float mx = logits[0];
#pragma unroll
    for (int a = 1; a < NANCH; ++a) mx = fmaxf(mx, logits[a]);
    float e[NANCH], s = 0.f;
#pragma unroll
    for (int a = 0; a < NANCH; ++a) { e[a] = __expf(logits[a] - mx); s += e[a]; }
    const float inv = 1.f / s;

    float4 o = xs;
#pragma unroll
    for (int a = 0; a < NANCH; ++a) {
        const float4 an = *(const float4*)(node_anchor + a * DIM + lane * 4);
        const float w = e[a] * inv;
        o.x += w * an.x; o.y += w * an.y; o.z += w * an.z; o.w += w * an.w;
    }
    *(float4*)(node_x + (size_t)node * DIM + lane * 4) = o;
}

// ---------------------------------------------------------------------------
// Kernel B: per-edge prompt + scatter-add into agg
// one wave per edge; x gathers mostly hit L2 (x = 51MB < 192MB L2)
// ---------------------------------------------------------------------------
__global__ __launch_bounds__(256) void edge_prompt_kernel(
    const float* __restrict__ x, const int* __restrict__ edge_index,
    const float* __restrict__ W_edge, const float* __restrict__ b_edge,
    const float* __restrict__ edge_anchor,
    float* __restrict__ edge_prompt, float* __restrict__ agg, int E)
{
    const int wave = threadIdx.x >> 5;
    const int lane = threadIdx.x & 31;
    const int e = blockIdx.x * 8 + wave;
    if (e >= E) return;

    const int src = edge_index[e];
    const int dst = edge_index[(size_t)E + e];

    const float4 xs = *(const float4*)(x + (size_t)src * DIM + lane * 4);
    const float4 xd = *(const float4*)(x + (size_t)dst * DIM + lane * 4);

    float logits[NANCH];
#pragma unroll
    for (int a = 0; a < NANCH; ++a) {
        const float4 ws = *(const float4*)(W_edge + a * 2 * DIM + lane * 4);
        const float4 wd = *(const float4*)(W_edge + a * 2 * DIM + DIM + lane * 4);
        float p = xs.x * ws.x + xs.y * ws.y + xs.z * ws.z + xs.w * ws.w
                + xd.x * wd.x + xd.y * wd.y + xd.z * wd.z + xd.w * wd.w;
        p = wave_reduce_sum(p) + b_edge[a];
        logits[a] = (p > 0.f) ? p : 0.01f * p;   // leaky_relu, slope 0.01
    }
    float mx = logits[0];
#pragma unroll
    for (int a = 1; a < NANCH; ++a) mx = fmaxf(mx, logits[a]);
    float w[NANCH], s = 0.f;
#pragma unroll
    for (int a = 0; a < NANCH; ++a) { w[a] = __expf(logits[a] - mx); s += w[a]; }
    const float inv = 1.f / s;

    float4 ep = make_float4(0.f, 0.f, 0.f, 0.f);
#pragma unroll
    for (int a = 0; a < NANCH; ++a) {
        const float4 an = *(const float4*)(edge_anchor + a * DIM + lane * 4);
        const float c = w[a] * inv;
        ep.x += c * an.x; ep.y += c * an.y; ep.z += c * an.z; ep.w += c * an.w;
    }
    *(float4*)(edge_prompt + (size_t)e * DIM + lane * 4) = ep;

    float* ps = agg + (size_t)src * DIM + lane * 4;
    float* pd = agg + (size_t)dst * DIM + lane * 4;
    __hip_atomic_fetch_add(ps + 0, ep.x, __ATOMIC_RELAXED, __HIP_MEMORY_SCOPE_AGENT);
    __hip_atomic_fetch_add(ps + 1, ep.y, __ATOMIC_RELAXED, __HIP_MEMORY_SCOPE_AGENT);
    __hip_atomic_fetch_add(ps + 2, ep.z, __ATOMIC_RELAXED, __HIP_MEMORY_SCOPE_AGENT);
    __hip_atomic_fetch_add(ps + 3, ep.w, __ATOMIC_RELAXED, __HIP_MEMORY_SCOPE_AGENT);
    __hip_atomic_fetch_add(pd + 0, ep.x, __ATOMIC_RELAXED, __HIP_MEMORY_SCOPE_AGENT);
    __hip_atomic_fetch_add(pd + 1, ep.y, __ATOMIC_RELAXED, __HIP_MEMORY_SCOPE_AGENT);
    __hip_atomic_fetch_add(pd + 2, ep.z, __ATOMIC_RELAXED, __HIP_MEMORY_SCOPE_AGENT);
    __hip_atomic_fetch_add(pd + 3, ep.w, __ATOMIC_RELAXED, __HIP_MEMORY_SCOPE_AGENT);
}

// ---------------------------------------------------------------------------
// Kernel D: fused agg_proj + 5-step ODE using v_wmma_f32_16x16x32_f16.
// Block = 16 nodes x 128 dims; 8 waves, each wave owns a 16x16 C tile.
// Tiles are staged with GLOBAL_LOAD_ASYNC_TO_LDS_B128 (ASYNCcnt) so the
// node_x DMA overlaps the agg_proj WMMA chain; weight B-fragments (W1a, W2)
// live in registers for the whole loop; ev/h tiles ping-pong through LDS.
// ---------------------------------------------------------------------------
__global__ __launch_bounds__(256) void ode_wmma_kernel(
    const float* __restrict__ node_x, const float* __restrict__ agg,
    const float* __restrict__ W1, const float* __restrict__ b1,
    const float* __restrict__ W2, const float* __restrict__ b2,
    float* __restrict__ ev_out, int N)
{
    __shared__ __align__(16) float     stageF32[16 * DIM];   // async DMA target
    __shared__ __align__(16) _Float16  tileA[16 * LSTR];
    __shared__ __align__(16) _Float16  tileB[16 * LSTR];

    const int tid   = threadIdx.x;
    const int wave  = tid >> 5;
    const int lane  = tid & 31;
    const int node0 = blockIdx.x * 16;
    if (node0 >= N) return;                    // uniform per block: EXEC stays full

    const int n0    = wave * 16;
    const int nc    = n0 + (lane & 15);        // output column for this lane (C layout)
    const int mbase = (lane < 16) ? 0 : 8;     // C-layout row base
    const int aoff  = (lane < 16) ? 0 : 16;    // A-frag K sub-offset (contiguous, matches B)
    const int koff  = (lane < 16) ? 0 : 16;    // B-frag K sub-offset (contiguous, matches A)

    const uint32_t stage_lds = (uint32_t)(uintptr_t)&stageF32[0]; // LDS byte offset

    // --- (1) async-stage agg tile (16x128 f32 = 8KB) into LDS
#pragma unroll
    for (int i = 0; i < 2; ++i) {
        const int elt = (i * 256 + tid) * 4;           // 4 floats = 16B / thread
        async_copy_b128(stage_lds + (uint32_t)elt * 4,
                        agg + (size_t)node0 * DIM + elt);
    }
    wait_async0();
    __syncthreads();

    // --- (2) convert agg tile f32 -> f16 into tileA (WMMA A layout source)
#pragma unroll
    for (int idx = tid; idx < 16 * DIM; idx += 256) {
        const int r = idx >> 7, c = idx & (DIM - 1);
        tileA[r * LSTR + c] = (_Float16)stageF32[idx];
    }
    __syncthreads();

    // --- (3) kick off node_x tile DMA; it overlaps the agg_proj WMMAs below
#pragma unroll
    for (int i = 0; i < 2; ++i) {
        const int elt = (i * 256 + tid) * 4;
        async_copy_b128(stage_lds + (uint32_t)elt * 4,
                        node_x + (size_t)node0 * DIM + elt);
    }

    // A-fragment loader: 16x32 f16 tile slice from LDS, two b128 loads/lane
    auto load_a = [&](const _Float16* t, int kk) -> v16h {
        const _Float16* p = t + (lane & 15) * LSTR + kk * 32 + aoff;
        const v8h lo = *(const v8h*)p;
        const v8h hi = *(const v8h*)(p + 8);
        v16h a;
#pragma unroll
        for (int i = 0; i < 8; ++i) { a[i] = lo[i]; a[i + 8] = hi[i]; }
        return a;
    };

    // --- (4) agg_proj = agg @ W1b^T + b1   (B[k][n] = W1b[n][k], contiguous k)
    v8f aggp = {};
    {
        v16h bw1b[4];
#pragma unroll
        for (int kk = 0; kk < 4; ++kk) {
            const float* r = W1 + (size_t)nc * (2 * DIM) + DIM + kk * 32 + koff;
#pragma unroll
            for (int i = 0; i < 16; ++i) bw1b[kk][i] = (_Float16)r[i];
        }
#pragma unroll
        for (int kk = 0; kk < 4; ++kk) {
            v16h a = load_a(tileA, kk);
            aggp = __builtin_amdgcn_wmma_f32_16x16x32_f16(
                false, a, false, bw1b[kk], (short)0, aggp, false, false);
        }
        const float bias1 = b1[nc];
#pragma unroll
        for (int j = 0; j < 8; ++j) aggp[j] += bias1;
    }

    // --- (5) node_x DMA complete + all waves done reading tileA
    wait_async0();
    __syncthreads();

    // --- (6) ev regs (f32, C layout) from LDS; convert node_x tile into tileA
    v8f ev;
#pragma unroll
    for (int j = 0; j < 8; ++j)
        ev[j] = stageF32[(mbase + j) * DIM + nc];
#pragma unroll
    for (int idx = tid; idx < 16 * DIM; idx += 256) {
        const int r = idx >> 7, c = idx & (DIM - 1);
        tileA[r * LSTR + c] = (_Float16)stageF32[idx];
    }
    __syncthreads();

    // --- (7) resident B fragments for the loop (loaded after agg_proj to
    //         keep peak VGPR pressure down): B[k][n] = W[n][k]
    v16h bw1[4], bw2[4];
#pragma unroll
    for (int kk = 0; kk < 4; ++kk) {
        const float* r1 = W1 + (size_t)nc * (2 * DIM) + kk * 32 + koff; // W1a
        const float* r2 = W2 + (size_t)nc * DIM + kk * 32 + koff;
#pragma unroll
        for (int i = 0; i < 16; ++i) {
            bw1[kk][i] = (_Float16)r1[i];
            bw2[kk][i] = (_Float16)r2[i];
        }
    }

    const float dt    = 1.0f / ODE_STEPS;
    const float bias2 = b2[nc];

#pragma unroll
    for (int step = 0; step < ODE_STEPS; ++step) {
        // h = relu(ev @ W1a^T + agg_proj)
        v8f acc = aggp;
#pragma unroll
        for (int kk = 0; kk < 4; ++kk) {
            v16h a = load_a(tileA, kk);
            acc = __builtin_amdgcn_wmma_f32_16x16x32_f16(
                false, a, false, bw1[kk], (short)0, acc, false, false);
        }
        // scatter h (f16) into tileB in C layout
#pragma unroll
        for (int j = 0; j < 8; ++j) {
            const float h = fmaxf(acc[j], 0.0f);
            tileB[(mbase + j) * LSTR + nc] = (_Float16)h;
        }
        __syncthreads();

        // ev += dt * (h @ W2^T + b2)
        v8f acc2 = {};
#pragma unroll
        for (int kk = 0; kk < 4; ++kk) {
            v16h a = load_a(tileB, kk);
            acc2 = __builtin_amdgcn_wmma_f32_16x16x32_f16(
                false, a, false, bw2[kk], (short)0, acc2, false, false);
        }
#pragma unroll
        for (int j = 0; j < 8; ++j) ev[j] += dt * (acc2[j] + bias2);

        if (step + 1 < ODE_STEPS) {
            __syncthreads();
#pragma unroll
            for (int j = 0; j < 8; ++j)
                tileA[(mbase + j) * LSTR + nc] = (_Float16)ev[j];
            __syncthreads();
        }
    }

    // --- write ev (16 consecutive n per half-wave -> coalesced 64B rows)
#pragma unroll
    for (int j = 0; j < 8; ++j)
        ev_out[(size_t)(node0 + mbase + j) * DIM + nc] = ev[j];
}

// ---------------------------------------------------------------------------
extern "C" void kernel_launch(void* const* d_in, const int* in_sizes, int n_in,
                              void* d_out, int out_size, void* d_ws, size_t ws_size,
                              hipStream_t stream)
{
    const float* x           = (const float*)d_in[0];
    const int*   edge_index  = (const int*)  d_in[1];
    const float* node_anchor = (const float*)d_in[2];
    const float* W_att       = (const float*)d_in[3];
    const float* b_att       = (const float*)d_in[4];
    const float* edge_anchor = (const float*)d_in[5];
    const float* W_edge      = (const float*)d_in[6];
    const float* b_edge      = (const float*)d_in[7];
    const float* W1          = (const float*)d_in[8];
    const float* b1          = (const float*)d_in[9];
    const float* W2          = (const float*)d_in[10];
    const float* b2          = (const float*)d_in[11];

    const int N = in_sizes[0] / DIM;
    const int E = in_sizes[1] / 2;

    float* ev_out = (float*)d_out;                    // (N, DIM)
    float* ep_out = ev_out + (size_t)N * DIM;         // (E, DIM)

    float* agg    = (float*)d_ws;                     // (N, DIM) scratch
    float* node_x = agg + (size_t)N * DIM;            // (N, DIM) scratch

    hipMemsetAsync(agg, 0, (size_t)N * DIM * sizeof(float), stream);

    node_prompt_kernel<<<(N + 7) / 8, 256, 0, stream>>>(
        x, W_att, b_att, node_anchor, node_x, N);

    edge_prompt_kernel<<<(E + 7) / 8, 256, 0, stream>>>(
        x, edge_index, W_edge, b_edge, edge_anchor, ep_out, agg, E);

    ode_wmma_kernel<<<(N + 15) / 16, 256, 0, stream>>>(
        node_x, agg, W1, b1, W2, b2, ev_out, N);
}
